// HierMoeNet_84121229460187
// MI455X (gfx1250) — matrix operand order repaired and checked by reading, not compiled, not run on hardware
//
#include <hip/hip_runtime.h>
#include <hip/hip_bf16.h>

// ---------------------------------------------------------------------------
// HierMoE on MI455X (gfx1250, wave32, WMMA):
//   GEMM1 (E x [512x1280 @ 1280x2560]) -> ReLU -> h (bf16 in d_ws)
//   GEMM2 (E x [512x2560 @ 2560x8])    -> logits (f32, d_out tail)
//   softmax(K=8) + tree path products  -> leaf_probs [512x73] (d_out head)
// fp32 inputs are converted to bf16 in-register and fed to
// v_wmma_f32_16x16x32_bf16 with fp32 accumulation.
// Register blocking: each wave owns 2 M-tiles x 4 N-tiles (32x64 outputs,
// 8 v8f accumulators) so every B fragment feeds two WMMAs.
// ---------------------------------------------------------------------------

typedef __attribute__((ext_vector_type(16))) __bf16 v16bf;
typedef __attribute__((ext_vector_type(8)))  float  v8f;

constexpr int Bsz  = 512;
constexpr int Fdim = 1280;
constexpr int Hdim = 2560;
constexpr int Kch  = 8;
constexpr int Nexp = 9;

static __device__ __forceinline__ v8f wmma_bf16(const v16bf& a, const v16bf& b, const v8f& c) {
    return __builtin_amdgcn_wmma_f32_16x16x32_bf16(
        /*neg_a=*/false, a, /*neg_b=*/false, b,
        /*c_mod=*/(short)0, c, /*reuse_a=*/false, /*reuse_b=*/false);
}

static __device__ __forceinline__ v16bf load_a_frag_f32(const float* __restrict__ p, int g) {
    // A fragment (16x32 bf16): lane L<16 -> row L, k in {0..7, 16..23};
    //                          lane L>=16 -> row L-16, k in {8..15, 24..31}.
    const float4 a0 = *(const float4*)(p + g * 8);
    const float4 a1 = *(const float4*)(p + g * 8 + 4);
    const float4 a2 = *(const float4*)(p + 16 + g * 8);
    const float4 a3 = *(const float4*)(p + 16 + g * 8 + 4);
    v16bf a;
    a[0]  = (__bf16)a0.x; a[1]  = (__bf16)a0.y; a[2]  = (__bf16)a0.z; a[3]  = (__bf16)a0.w;
    a[4]  = (__bf16)a1.x; a[5]  = (__bf16)a1.y; a[6]  = (__bf16)a1.z; a[7]  = (__bf16)a1.w;
    a[8]  = (__bf16)a2.x; a[9]  = (__bf16)a2.y; a[10] = (__bf16)a2.z; a[11] = (__bf16)a2.w;
    a[12] = (__bf16)a3.x; a[13] = (__bf16)a3.y; a[14] = (__bf16)a3.z; a[15] = (__bf16)a3.w;
    return a;
}

// ---------------------------------------------------------------------------
// Kernel 1: h[e,b,n] = relu( sum_f x[b,f] * W1[e,f,n] + b1[e,n] ), bf16 out.
// Block = 256 threads (8 waves). Wave tile: 32(M) x 64(N); block: 64 x 256.
// B fragment (32x16 bf16): column n = lane%16; lanes 0..15 hold rows {0..7, 16..23};
//                          lanes 16..31 hold rows {8..15, 24..31} (same k order as A).
// ---------------------------------------------------------------------------
__global__ __launch_bounds__(256) void gemm1_relu_kernel(
    const float* __restrict__ x, const float* __restrict__ W1,
    const float* __restrict__ b1, __bf16* __restrict__ hbuf)
{
    const int lane = threadIdx.x & 31;
    const int wave = threadIdx.x >> 5;
    const int g = lane >> 4;    // half-wave group
    const int r = lane & 15;

    constexpr int nBlkM = Bsz / 64;    // 8
    constexpr int nBlkN = Hdim / 256;  // 10
    const int bid = blockIdx.x;
    const int e   = bid / (nBlkM * nBlkN);
    const int rem = bid % (nBlkM * nBlkN);
    const int nb  = rem / nBlkM;       // n-slab (m varies fastest -> L2 reuse of W1)
    const int mbk = rem % nBlkM;

    const int m0 = mbk * 64 + (wave & 1) * 32;          // wave's 32 batch rows
    const int n0 = nb * 256 + (wave >> 1) * 64;         // wave's 64 h columns

    const float* __restrict__ xrow0 = x + (size_t)(m0 + r) * Fdim;
    const float* __restrict__ xrow1 = x + (size_t)(m0 + 16 + r) * Fdim;
    const float* __restrict__ W1e   = W1 + (size_t)e * Fdim * Hdim;

    v8f acc[2][4] = {};

    for (int kk = 0; kk < Fdim; kk += 32) {
        // ---- A fragments: two 16-row tiles, contiguous f32 -> bf16 ----
        const v16bf afrag0 = load_a_frag_f32(xrow0 + kk, g);
        const v16bf afrag1 = load_a_frag_f32(xrow1 + kk, g);

        const float* __restrict__ wbase =
            W1e + (size_t)(kk + g * 8) * Hdim + n0 + r;

        // speculative prefetch of next W1 k-tile (global_prefetch_b8)
        if (kk + 32 < Fdim)
            __builtin_prefetch(wbase + (size_t)32 * Hdim, 0, 1);

        #pragma unroll
        for (int nt = 0; nt < 4; ++nt) {
            const float* __restrict__ wp = wbase + nt * 16;
            v16bf bfrag;
            #pragma unroll
            for (int t = 0; t < 16; ++t)
                bfrag[t] = (__bf16)wp[(size_t)(t < 8 ? t : t + 8) * Hdim];   // coalesced across lanes
            acc[0][nt] = wmma_bf16(afrag0, bfrag, acc[0][nt]);
            acc[1][nt] = wmma_bf16(afrag1, bfrag, acc[1][nt]);
        }
    }

    // ---- bias + ReLU + store bf16 ----
    #pragma unroll
    for (int nt = 0; nt < 4; ++nt) {
        const int col = n0 + nt * 16 + r;
        const float bias = b1[e * Hdim + col];
        #pragma unroll
        for (int mt = 0; mt < 2; ++mt) {
            #pragma unroll
            for (int i = 0; i < 8; ++i) {
                const int row = m0 + mt * 16 + i + g * 8;  // C layout: lanes16-31 -> M+8
                float v = acc[mt][nt][i] + bias;
                v = v > 0.0f ? v : 0.0f;
                hbuf[((size_t)e * Bsz + row) * Hdim + col] = (__bf16)v;
            }
        }
    }
}

// ---------------------------------------------------------------------------
// Kernel 2: logits[e,b,k] = sum_h h[e,b,h] * W2[e,h,k] + b2[e,k]
// One wave per 16-batch tile; one WMMA tile, N columns >= 8 are clamped
// duplicates (never stored).
// ---------------------------------------------------------------------------
__global__ __launch_bounds__(256) void gemm2_logits_kernel(
    const __bf16* __restrict__ hbuf, const float* __restrict__ W2,
    const float* __restrict__ b2, float* __restrict__ logits)
{
    const int lane = threadIdx.x & 31;
    const int wave = threadIdx.x >> 5;
    const int g = lane >> 4;
    const int r = lane & 15;

    const int e  = blockIdx.x >> 2;              // 9 experts * 4 blocks
    const int bb = blockIdx.x & 3;
    const int b0 = (bb * 8 + wave) * 16;

    const float*  __restrict__ W2e  = W2 + (size_t)e * Hdim * Kch;
    const __bf16* __restrict__ hrow = hbuf + ((size_t)e * Bsz + b0 + r) * Hdim;
    const int nc = r < Kch ? r : (Kch - 1);      // clamp: avoid OOB, cols>=8 dead

    v8f acc = {};

    for (int kk = 0; kk < Hdim; kk += 32) {
        v16bf afrag;
        const __bf16* __restrict__ ap0 = hrow + kk + g * 8;
        const __bf16* __restrict__ ap1 = hrow + kk + 16 + g * 8;
        #pragma unroll
        for (int t = 0; t < 8; ++t) afrag[t]     = ap0[t];   // contiguous b128
        #pragma unroll
        for (int t = 0; t < 8; ++t) afrag[8 + t] = ap1[t];   // contiguous b128

        v16bf bfrag;
        const float* __restrict__ wp = W2e + (size_t)(kk + g * 8) * Kch + nc;
        #pragma unroll
        for (int t = 0; t < 16; ++t)
            bfrag[t] = (__bf16)wp[(t < 8 ? t : t + 8) * Kch];

        acc = wmma_bf16(afrag, bfrag, acc);
    }

    if (r < Kch) {
        const float bias = b2[e * Kch + r];
        #pragma unroll
        for (int i = 0; i < 8; ++i) {
            const int row = b0 + i + g * 8;
            logits[((size_t)e * Bsz + row) * Kch + r] = acc[i] + bias;
        }
    }
}

// ---------------------------------------------------------------------------
// Kernel 3: per batch row: softmax over K=8 for all 9 experts, then
// leaf[b] = [1, p1(8), p1_i * p2_ij (64)]  -> [B, 73]
// ---------------------------------------------------------------------------
__global__ __launch_bounds__(256) void softmax_leaf_kernel(
    const float* __restrict__ logits, float* __restrict__ leaf)
{
    const int b = blockIdx.x * blockDim.x + threadIdx.x;
    if (b >= Bsz) return;

    float sm[Nexp][Kch];
    #pragma unroll
    for (int e = 0; e < Nexp; ++e) {
        const float* lp = logits + ((size_t)e * Bsz + b) * Kch;
        float m = lp[0];
        #pragma unroll
        for (int k = 1; k < Kch; ++k) m = fmaxf(m, lp[k]);
        float s = 0.0f;
        #pragma unroll
        for (int k = 0; k < Kch; ++k) { float t = __expf(lp[k] - m); sm[e][k] = t; s += t; }
        const float inv = 1.0f / s;
        #pragma unroll
        for (int k = 0; k < Kch; ++k) sm[e][k] *= inv;
    }

    float* out = leaf + (size_t)b * 73;
    out[0] = 1.0f;
    #pragma unroll
    for (int i = 0; i < Kch; ++i) out[1 + i] = sm[0][i];
    #pragma unroll
    for (int i = 0; i < Kch; ++i)
        #pragma unroll
        for (int j = 0; j < Kch; ++j)
            out[9 + i * Kch + j] = sm[0][i] * sm[1 + i][j];
}

// ---------------------------------------------------------------------------
extern "C" void kernel_launch(void* const* d_in, const int* in_sizes, int n_in,
                              void* d_out, int out_size, void* d_ws, size_t ws_size,
                              hipStream_t stream) {
    const float* x  = (const float*)d_in[0];   // [B,F]
    const float* W1 = (const float*)d_in[1];   // [E,F,H]
    const float* b1 = (const float*)d_in[2];   // [E,H]
    const float* W2 = (const float*)d_in[3];   // [E,H,K]
    const float* b2 = (const float*)d_in[4];   // [E,K]

    float* leaf   = (float*)d_out;                        // [B,73]
    float* logits = (float*)d_out + (size_t)Bsz * 73;     // [E,B,K]

    __bf16* hbuf = (__bf16*)d_ws;   // [E,B,H] bf16 = 23.6 MB scratch

    // GEMM1 + ReLU: 9 experts * (512/64) m-blocks * (2560/256) n-slabs
    const int grid1 = Nexp * (Bsz / 64) * (Hdim / 256);   // 720
    gemm1_relu_kernel<<<grid1, 256, 0, stream>>>(x, W1, b1, hbuf);

    // GEMM2: 9 experts * 4 blocks (8 waves = 128 batch rows each)
    gemm2_logits_kernel<<<Nexp * 4, 256, 0, stream>>>(hbuf, W2, b2, logits);

    // Softmax + tree path products
    softmax_leaf_kernel<<<(Bsz + 255) / 256, 256, 0, stream>>>(logits, leaf);
}
